// Attention_67216238182661
// MI455X (gfx1250) — matrix operand order repaired and checked by reading, not compiled, not run on hardware
//
#include <hip/hip_runtime.h>
#include <hip/hip_bf16.h>

// Problem constants (match reference)
#define BB 2
#define HH 4
#define TT 2048
#define NN 2048
#define DD 256

typedef __attribute__((ext_vector_type(16))) __bf16        v16bf;
typedef __attribute__((ext_vector_type(8)))  float         v8f;
typedef __attribute__((ext_vector_type(4)))  unsigned int  v4u;

// Tiling
#define QROWS 128      // query rows per block (8 waves x 16)
#define KKEYS 64       // key rows per tile
#define NCHUNK 128     // feature-dim chunk staged in LDS
#define NCHUNKS (NN / NCHUNK)   // 16
#define NWAVES 8

// Padded LDS row strides (bf16 elems): +16B per row kills bank conflicts
#define QQP (NCHUNK + 8)   // 136 -> 272B row stride
#define QKP (NCHUNK + 8)
#define VTP (KKEYS + 8)    // 72  -> 144B row stride
#define STP (KKEYS + 8)

#define QQ_ELEMS (QROWS * QQP)        // 17408
#define QK_ELEMS (KKEYS * QKP)        //  8704
#define VT_ELEMS (DD * VTP)           // 18432
#define ST_ELEMS (NWAVES * 16 * STP)  //  9216

// Element offsets within smem (regions: qq0,qq1,qk0,qk1,vt,st)
#define QK_BASE_E (2 * QQ_ELEMS)
#define VT_BASE_E (2 * QQ_ELEMS + 2 * QK_ELEMS)
#define ST_BASE_E (VT_BASE_E + VT_ELEMS)
#define LDS_BYTES ((size_t)(ST_BASE_E + ST_ELEMS) * 2u)   // 159744 (~156KB)

// ---------------------------------------------------------------------------
// CDNA5 async global->LDS copy (ASYNCcnt-tracked, no VGPR data round trip)
// ---------------------------------------------------------------------------
__device__ __forceinline__ void async_copy_b128(unsigned lds_byte_off, const void* gaddr) {
    asm volatile("global_load_async_to_lds_b128 %0, %1, off"
                 :
                 : "v"(lds_byte_off), "v"((unsigned long long)(size_t)gaddr)
                 : "memory");
}
__device__ __forceinline__ void wait_async0() {
    asm volatile("s_wait_asynccnt 0x0" ::: "memory");
}

// WMMA operand fragment: two 16B LDS loads per lane
union Frag { v4u q[2]; v16bf v; };

// A fragment (16x32 bf16): lane runs at bytes [16*half,+16) and [32+16*half,+16)
__device__ __forceinline__ void load_afrag(Frag& f, const __bf16* base, int half) {
    const v4u* a4 = (const v4u*)base;
    f.q[0] = a4[half];
    f.q[1] = a4[2 + half];
}
// B fragment (32x16 bf16): lane runs at bytes [16*half,+16) and [32+16*half,+16)
__device__ __forceinline__ void load_bfrag(Frag& f, const __bf16* base, int half) {
    const v4u* b4 = (const v4u*)base;
    f.q[0] = b4[half];
    f.q[1] = b4[2 + half];
}

// ---------------------------------------------------------------------------
// Pre-pass 1: RoPE(Q) -> bf16, one thread per (even,odd) pair
// ---------------------------------------------------------------------------
__global__ void __launch_bounds__(256) rope_to_bf16_kernel(
    const float* __restrict__ Q, unsigned int* __restrict__ QRpacked)
{
    const size_t total_pairs = (size_t)BB * HH * TT * (NN / 2);
    size_t p = (size_t)blockIdx.x * blockDim.x + threadIdx.x;
    if (p >= total_pairs) return;

    const int n2 = (int)(p % (NN / 2));
    const int t  = (int)((p / (NN / 2)) % TT);

    const float f  = exp2f(-16.0f * (float)(2 * n2) * (1.0f / (float)NN)) * 0.15915494309189535f;
    float ph = (float)t * f;
    ph = (ph - floorf(ph)) * 6.283185307179586f;
    float s, c;
    __sincosf(ph, &s, &c);

    const float2 q = ((const float2*)Q)[p];
    const float r0 = q.x * c - q.y * s;
    const float r1 = q.y * c + q.x * s;

    const __bf16 h0 = (__bf16)r0;
    const __bf16 h1 = (__bf16)r1;
    QRpacked[p] = (unsigned int)__builtin_bit_cast(unsigned short, h0)
                | ((unsigned int)__builtin_bit_cast(unsigned short, h1) << 16);
}

// ---------------------------------------------------------------------------
// Pre-pass 2: V fp32 -> bf16
// ---------------------------------------------------------------------------
__global__ void __launch_bounds__(256) v_to_bf16_kernel(
    const float* __restrict__ V, unsigned int* __restrict__ Vpacked)
{
    const size_t total_pairs = (size_t)BB * TT * (DD / 2);
    size_t p = (size_t)blockIdx.x * blockDim.x + threadIdx.x;
    if (p >= total_pairs) return;
    const float2 v = ((const float2*)V)[p];
    const __bf16 h0 = (__bf16)v.x;
    const __bf16 h1 = (__bf16)v.y;
    Vpacked[p] = (unsigned int)__builtin_bit_cast(unsigned short, h0)
               | ((unsigned int)__builtin_bit_cast(unsigned short, h1) << 16);
}

// ---------------------------------------------------------------------------
// Stage one 128-wide feature chunk (query rows + key rows) into buffer `buf`
// with async b128 copies (12 per thread).
// ---------------------------------------------------------------------------
__device__ __forceinline__ void stage_chunk(const __bf16* __restrict__ QRbh,
                                            int qbase, int s0, int n0,
                                            int buf, int tid)
{
    const unsigned qqoff = (unsigned)buf * (QQ_ELEMS * 2u);
    const unsigned qkoff = (unsigned)(QK_BASE_E * 2) + (unsigned)buf * (QK_ELEMS * 2u);
    for (int e4 = tid; e4 < QROWS * (NCHUNK / 8); e4 += 256) {
        const int r = e4 >> 4, c8 = (e4 & 15) * 8;
        async_copy_b128(qqoff + (unsigned)(r * QQP + c8) * 2u,
                        QRbh + (size_t)(qbase + r) * NN + n0 + c8);
    }
    for (int e4 = tid; e4 < KKEYS * (NCHUNK / 8); e4 += 256) {
        const int r = e4 >> 4, c8 = (e4 & 15) * 8;
        async_copy_b128(qkoff + (unsigned)(r * QKP + c8) * 2u,
                        QRbh + (size_t)(s0 + r) * NN + n0 + c8);
    }
}

// ---------------------------------------------------------------------------
// Main kernel: masked Gram-attention, flash-style.
// grid = (T/QROWS, B*H), block = 256 threads (8 waves, wave32).
// ---------------------------------------------------------------------------
__global__ void __launch_bounds__(256, 1) bdh_attn_kernel(
    const __bf16* __restrict__ QR,   // (B,H,T,N) bf16, roped
    const __bf16* __restrict__ Vb,   // (B,T,D) bf16
    float* __restrict__ Out)         // (B,H,T,D) fp32
{
    extern __shared__ __bf16 smem[];

    const int tid   = threadIdx.x;
    const int w     = tid >> 5;
    const int lane  = tid & 31;
    const int m16   = lane & 15;
    const int half  = lane >> 4;
    const int bh    = blockIdx.y;
    const int b     = bh / HH;
    const int qbase = blockIdx.x * QROWS;
    const int qw    = qbase + w * 16;

    const __bf16* QRbh = QR + (size_t)bh * TT * NN;
    const __bf16* Vbb  = Vb + (size_t)b * TT * DD;

    v8f o_acc[16];
#pragma unroll
    for (int i = 0; i < 16; ++i) {
#pragma unroll
        for (int j = 0; j < 8; ++j) o_acc[i][j] = 0.0f;
    }

    const int nkt = (qbase + QROWS - 2) / KKEYS + 1;

    for (int kt = 0; kt < nkt; ++kt) {
        const int s0 = kt * KKEYS;
        const bool live = (s0 <= qw + 14);

        __syncthreads();   // previous tile's S@V (vt readers) and last-chunk compute done

        // V tile transposed into LDS: vt[d][key] = V[s0+key][d]
        for (int e = tid; e < KKEYS * (DD / 2); e += 256) {
            const int key = e >> 7, dp = e & 127;
            const unsigned pack =
                *(const unsigned int*)(Vbb + (size_t)(s0 + key) * DD + 2 * dp);
            smem[VT_BASE_E + (2 * dp) * VTP + key] =
                __builtin_bit_cast(__bf16, (unsigned short)(pack & 0xFFFFu));
            smem[VT_BASE_E + (2 * dp + 1) * VTP + key] =
                __builtin_bit_cast(__bf16, (unsigned short)(pack >> 16));
        }

        v8f s_acc[4];
#pragma unroll
        for (int j = 0; j < 4; ++j) {
#pragma unroll
            for (int v = 0; v < 8; ++v) s_acc[j][v] = 0.0f;
        }

        // Prologue: stage chunk 0 into buffer 0
        stage_chunk(QRbh, qbase, s0, 0, 0, tid);

        for (int i = 0; i < NCHUNKS; ++i) {
            const int cur = i & 1;
            wait_async0();       // this wave's staged chunk i complete in LDS
            __syncthreads();     // all waves' chunk i visible; chunk-(i-1) readers done
            if (i + 1 < NCHUNKS)
                stage_chunk(QRbh, qbase, s0, (i + 1) * NCHUNK, cur ^ 1, tid);

            if (live) {
                // LDS bases as integer offsets from smem (keeps DS address space)
                const __bf16* qqc = smem + cur * QQ_ELEMS;
                const __bf16* qkc = smem + QK_BASE_E + cur * QK_ELEMS;
                const __bf16* arow = qqc + (w * 16 + m16) * QQP;

                // Register double-buffered fragment pipeline over kk
                Frag af[2]; Frag bf[2][4];
                load_afrag(af[0], arow, half);
#pragma unroll
                for (int j = 0; j < 4; ++j)
                    load_bfrag(bf[0][j], qkc + (j * 16 + m16) * QKP, half);

#pragma unroll
                for (int kki = 0; kki < 4; ++kki) {
                    const int curf = kki & 1, nxtf = curf ^ 1;
                    if (kki < 3) {
                        const int kk = (kki + 1) * 32;
                        load_afrag(af[nxtf], arow + kk, half);
#pragma unroll
                        for (int j = 0; j < 4; ++j)
                            load_bfrag(bf[nxtf][j], qkc + (j * 16 + m16) * QKP + kk, half);
                        // Pin schedule: [10 DS reads (iter kki+1)] then [4 WMMA (iter kki)]
                        __builtin_amdgcn_sched_group_barrier(0x100, 10, 0);
                    }
                    __builtin_amdgcn_sched_group_barrier(0x008, 4, 0);
#pragma unroll
                    for (int j = 0; j < 4; ++j)
                        s_acc[j] = __builtin_amdgcn_wmma_f32_16x16x32_bf16(
                            false, af[curf].v, false, bf[curf][j].v, (short)0,
                            s_acc[j], false, false);
                }
            }
        }

        if (live) {
            // Mask (keep s < q), convert to bf16, park in per-wave LDS scratch
            const int swbase = ST_BASE_E + w * 16 * STP;
#pragma unroll
            for (int j = 0; j < 4; ++j) {
                const int k  = j * 16 + m16;
                const int sg = s0 + k;
#pragma unroll
                for (int v = 0; v < 8; ++v) {
                    const int m = v + (half << 3);
                    const float val = (sg < qw + m) ? s_acc[j][v] : 0.0f;
                    smem[swbase + m * STP + k] = (__bf16)val;
                }
            }
            // Same-wave LDS RAW kept in order by DScnt.

            Frag sa[2];
#pragma unroll
            for (int h2 = 0; h2 < 2; ++h2)
                load_afrag(sa[h2], smem + swbase + m16 * STP + h2 * 32, half);

            // O += S @ V : d-tile pairs, register double-buffered B fragments
            const __bf16* vt = smem + VT_BASE_E;
            Frag vf[2][4];
#pragma unroll
            for (int h2 = 0; h2 < 2; ++h2) {
#pragma unroll
                for (int q2 = 0; q2 < 2; ++q2)
                    load_bfrag(vf[0][q2 * 2 + h2], vt + (q2 * 16 + m16) * VTP + h2 * 32, half);
            }
#pragma unroll
            for (int dtp = 0; dtp < 8; ++dtp) {
                const int curf = dtp & 1, nxtf = curf ^ 1;
                if (dtp < 7) {
                    const int dtn = (dtp + 1) * 2;
#pragma unroll
                    for (int q2 = 0; q2 < 2; ++q2) {
#pragma unroll
                        for (int h2 = 0; h2 < 2; ++h2)
                            load_bfrag(vf[nxtf][q2 * 2 + h2],
                                       vt + ((dtn + q2) * 16 + m16) * VTP + h2 * 32, half);
                    }
                    __builtin_amdgcn_sched_group_barrier(0x100, 8, 0);
                }
                __builtin_amdgcn_sched_group_barrier(0x008, 4, 0);
                const int dt = dtp * 2;
#pragma unroll
                for (int q2 = 0; q2 < 2; ++q2) {
#pragma unroll
                    for (int h2 = 0; h2 < 2; ++h2)
                        o_acc[dt + q2] = __builtin_amdgcn_wmma_f32_16x16x32_bf16(
                            false, sa[h2].v, false, vf[curf][q2 * 2 + h2].v, (short)0,
                            o_acc[dt + q2], false, false);
                }
            }
        }
    }

    // Write O: Out[(bh*T + qw + m)*D + dt*16 + m16]
    float* obase = Out + ((size_t)bh * TT + qw) * DD;
#pragma unroll
    for (int dt = 0; dt < 16; ++dt) {
#pragma unroll
        for (int v = 0; v < 8; ++v) {
            const int m = v + (half << 3);
            obase[(size_t)m * DD + dt * 16 + m16] = o_acc[dt][v];
        }
    }
}

// ---------------------------------------------------------------------------
// Host launcher. d_ws: bf16 QR (B*H*T*N) then bf16 V (B*T*D) = 69,206,016 B.
// ---------------------------------------------------------------------------
extern "C" void kernel_launch(void* const* d_in, const int* in_sizes, int n_in,
                              void* d_out, int out_size, void* d_ws, size_t ws_size,
                              hipStream_t stream)
{
    (void)in_sizes; (void)n_in; (void)out_size; (void)ws_size;
    const float* Q = (const float*)d_in[0];
    const float* V = (const float*)d_in[1];
    float* Out = (float*)d_out;

    __bf16* QRb = (__bf16*)d_ws;
    __bf16* Vb  = QRb + (size_t)BB * HH * TT * NN;

    {
        const size_t pairs = (size_t)BB * HH * TT * (NN / 2);
        rope_to_bf16_kernel<<<(unsigned)((pairs + 255) / 256), 256, 0, stream>>>(
            Q, (unsigned int*)QRb);
    }
    {
        const size_t pairs = (size_t)BB * TT * (DD / 2);
        v_to_bf16_kernel<<<(unsigned)((pairs + 255) / 256), 256, 0, stream>>>(
            V, (unsigned int*)Vb);
    }
    {
        dim3 grid(TT / QROWS, BB * HH);
        bdh_attn_kernel<<<grid, 256, LDS_BYTES, stream>>>(QRb, Vb, Out);
    }
}
